// KronQRInjectedLinear_QR_88596585382844
// MI455X (gfx1250) — compile-verified, matrix operand order checked by reading, not run on hardware
//
#include <hip/hip_runtime.h>
#include <hip/hip_bf16.h>

// CDNA5 / gfx1250 split-bf16 GEMM pipeline for KronQRInjectedLinear.
//   GEMM1: QR  = Q @ (R + triu(lora loraT))      [B-operand fused]
//   GEMM2: Wt  = KronExpand(kron1,kron2) @ QR    [A-operand generated on the fly]
//   GEMM3: out = x @ Wt
// v_wmma_f32_16x16x32_bf16 with hi/lo fp32->bf16 splitting (3 WMMA products per
// K=32 step, fp32 accumulation, ~2^-16 relative error). Register double-buffered
// global->LDS staging + global_prefetch of tile k+2 to hide memory latency.

typedef __attribute__((ext_vector_type(16))) __bf16 v16bf;
typedef __attribute__((ext_vector_type(8)))  float  v8f;

union Frag16 { v16bf v; uint4 q[2]; };

constexpr int TBM  = 128;   // block tile M
constexpr int TBN  = 128;   // block tile N
constexpr int TBK  = 32;    // block tile K (one WMMA K)
constexpr int LDSK = 40;    // 32 + 8 pad: 80B row stride -> 16B aligned, conflict-free

enum { AMODE_DIRECT = 0, AMODE_KRON = 1 };
enum { BMODE_DIRECT = 0, BMODE_TRIU = 1 };

template <int AMODE, int BMODE>
__global__ __launch_bounds__(256)
void gemm_bf16x2(const float* __restrict__ A,
                 const float* __restrict__ B,
                 float* __restrict__ C,
                 int M, int N, int K,
                 const float* __restrict__ kron1,
                 const float* __restrict__ kron2,
                 const float* __restrict__ lora)
{
  __shared__ __bf16 sAh[TBM][LDSK];   // A tile, hi bf16 plane  [m][k]
  __shared__ __bf16 sAl[TBM][LDSK];   // A tile, lo bf16 plane
  __shared__ __bf16 sBh[TBN][LDSK];   // B tile transposed      [n][k]
  __shared__ __bf16 sBl[TBN][LDSK];

  const int tid   = threadIdx.x;
  const int lane  = tid & 31;
  const int wave  = tid >> 5;
  const int waveM = wave >> 2;        // 0..1 -> 64 rows each
  const int waveN = wave & 3;         // 0..3 -> 32 cols each
  const int half  = lane >> 4;        // lane group (0: lanes 0-15, 1: 16-31)
  const int l16   = lane & 15;

  const int mBlock = blockIdx.y * TBM;
  const int nBlock = blockIdx.x * TBN;

  v8f acc[4][2] = {};

  // staging assignments (256 threads, 16 floats each per tile)
  const int ar = tid >> 1;            // A tile row   0..127
  const int ac = (tid & 1) * 16;      // A tile col   0 or 16
  const int bk = tid & 31;            // B tile k-row 0..31
  const int bc = (tid >> 5) * 16;     // B tile col   0..112 (step 16)

  float ra[16];                       // double-buffer regs: next A tile
  float rb[16];                       //                     next B tile

  auto loadA = [&](float* v, int k0) {
    const int grow = mBlock + ar;
    if constexpr (AMODE == AMODE_DIRECT) {
      const float4* src = (const float4*)(A + (size_t)grow * K + k0 + ac);
      #pragma unroll
      for (int j = 0; j < 4; ++j) {
        float4 t = src[j];
        v[4*j+0] = t.x; v[4*j+1] = t.y; v[4*j+2] = t.z; v[4*j+3] = t.w;
      }
    } else {
      // A_eff[r,c] = kron1[r>>1, c>>1] * kron2[r&1, c&1]
      const int   Kh  = K >> 1;
      const float k2a = kron2[(grow & 1) * 2 + 0];
      const float k2b = kron2[(grow & 1) * 2 + 1];
      const float* k1 = kron1 + (size_t)(grow >> 1) * Kh + ((k0 + ac) >> 1);
      #pragma unroll
      for (int j = 0; j < 8; ++j) {
        float kv = k1[j];
        v[2*j+0] = kv * k2a;
        v[2*j+1] = kv * k2b;
      }
    }
  };

  auto loadB = [&](float* v, int k0) {
    const int gk = k0 + bk;
    const int gc = nBlock + bc;
    const float4* src = (const float4*)(B + (size_t)gk * N + gc);
    #pragma unroll
    for (int j = 0; j < 4; ++j) {
      float4 t = src[j];
      v[4*j+0] = t.x; v[4*j+1] = t.y; v[4*j+2] = t.z; v[4*j+3] = t.w;
    }
    if constexpr (BMODE == BMODE_TRIU) {
      // B_eff[k,n] = R[k,n] + (n>=k) * (lora[k,:] . lora[n,:])
      const float la0 = lora[gk*2 + 0];
      const float la1 = lora[gk*2 + 1];
      #pragma unroll
      for (int j = 0; j < 16; ++j) {
        int   n  = gc + j;
        float up = la0 * lora[n*2 + 0] + la1 * lora[n*2 + 1];
        v[j] += (n >= gk) ? up : 0.0f;
      }
    }
  };

  auto storeTiles = [&](const float* va, const float* vb) {
    #pragma unroll
    for (int j = 0; j < 16; ++j) {
      float  u = va[j];
      __bf16 h = (__bf16)u;
      sAh[ar][ac + j] = h;
      sAl[ar][ac + j] = (__bf16)(u - (float)h);
    }
    #pragma unroll
    for (int j = 0; j < 16; ++j) {
      float  u = vb[j];
      __bf16 h = (__bf16)u;
      sBh[bc + j][bk] = h;
      sBl[bc + j][bk] = (__bf16)(u - (float)h);
    }
  };

  // prologue: fetch tile 0 into registers
  loadA(ra, 0);
  loadB(rb, 0);

  for (int k0 = 0; k0 < K; k0 += TBK) {
    // commit current tile (convert fp32 -> hi/lo bf16 planes) to LDS
    storeTiles(ra, rb);
    __syncthreads();

    // issue global loads for tile k+1: latency overlaps the WMMAs below
    const int kn = k0 + TBK;
    if (kn < K) {
      loadA(ra, kn);
      loadB(rb, kn);
    }
    // warm L2 for tile k+2 (global_prefetch_b8)
    const int kp = kn + TBK;
    if (kp < K) {
      if constexpr (AMODE == AMODE_DIRECT)
        __builtin_prefetch(A + (size_t)(mBlock + ar) * K + kp + ac, 0, 1);
      __builtin_prefetch(B + (size_t)(kp + bk) * N + nBlock + bc, 0, 1);
    }

    // ---------------- fragments from LDS ----------------
    // 16-bit A layout (ISA 7.12.2): lane half h, element e -> K = (e&8)*2 + h*8 + (e&7)
    // => two contiguous 16B chunks at columns [h*8, h*8+8) and [16+h*8, 16+h*8+8).
    Frag16 afh[4], afl[4], bfh[2], bfl[2];
    #pragma unroll
    for (int mt = 0; mt < 4; ++mt) {
      int r = waveM*64 + mt*16 + l16;
      afh[mt].q[0] = *(const uint4*)&sAh[r][half*8];
      afh[mt].q[1] = *(const uint4*)&sAh[r][16 + half*8];
      afl[mt].q[0] = *(const uint4*)&sAl[r][half*8];
      afl[mt].q[1] = *(const uint4*)&sAl[r][16 + half*8];
    }
    #pragma unroll
    for (int nt = 0; nt < 2; ++nt) {
      int r = waveN*32 + nt*16 + l16;
      bfh[nt].q[0] = *(const uint4*)&sBh[r][half*8];
      bfh[nt].q[1] = *(const uint4*)&sBh[r][16 + half*8];
      bfl[nt].q[0] = *(const uint4*)&sBl[r][half*8];
      bfl[nt].q[1] = *(const uint4*)&sBl[r][16 + half*8];
    }

    // ---------------- 3-product split-bf16 MACs ----------------
    #pragma unroll
    for (int mt = 0; mt < 4; ++mt) {
      #pragma unroll
      for (int nt = 0; nt < 2; ++nt) {
        acc[mt][nt] = __builtin_amdgcn_wmma_f32_16x16x32_bf16(
            false, afh[mt].v, false, bfh[nt].v, (short)0, acc[mt][nt], false, false);
        acc[mt][nt] = __builtin_amdgcn_wmma_f32_16x16x32_bf16(
            false, afh[mt].v, false, bfl[nt].v, (short)0, acc[mt][nt], false, false);
        acc[mt][nt] = __builtin_amdgcn_wmma_f32_16x16x32_bf16(
            false, afl[mt].v, false, bfh[nt].v, (short)0, acc[mt][nt], false, false);
      }
    }
    __syncthreads();
  }

  // ---------------- store (C/D layout: VGPR v -> row v + 8*half, col = lane&15) ----------------
  #pragma unroll
  for (int mt = 0; mt < 4; ++mt) {
    #pragma unroll
    for (int nt = 0; nt < 2; ++nt) {
      int row0 = mBlock + waveM*64 + mt*16 + half*8;
      int col  = nBlock + waveN*32 + nt*16 + l16;
      #pragma unroll
      for (int v = 0; v < 8; ++v) {
        C[(size_t)(row0 + v) * N + col] = acc[mt][nt][v];
      }
    }
  }
}

extern "C" void kernel_launch(void* const* d_in, const int* in_sizes, int n_in,
                              void* d_out, int out_size, void* d_ws, size_t ws_size,
                              hipStream_t stream) {
  (void)in_sizes; (void)n_in; (void)out_size; (void)ws_size;
  const float* x     = (const float*)d_in[0];   // [4,2048,4096] -> 8192 x 4096
  const float* Q     = (const float*)d_in[1];   // [4096,4096]
  const float* R     = (const float*)d_in[2];   // [4096,4096]
  const float* lora  = (const float*)d_in[3];   // [4096,2]
  const float* kron1 = (const float*)d_in[4];   // [2048,2048]
  const float* kron2 = (const float*)d_in[5];   // [2,2]
  float* out = (float*)d_out;

  constexpr size_t MAT = (size_t)4096 * 4096;
  float* QRbuf = (float*)d_ws;        // 64 MB
  float* Wtbuf = QRbuf + MAT;         // 64 MB (total ws use: 128 MB)

  dim3 blk(256);
  dim3 g1(4096 / TBN, 4096 / TBM);
  dim3 g3(4096 / TBN, 8192 / TBM);

  // 1) QR = Q @ (R + triu(lora loraT))
  hipLaunchKernelGGL((gemm_bf16x2<AMODE_DIRECT, BMODE_TRIU>), g1, blk, 0, stream,
                     Q, R, QRbuf, 4096, 4096, 4096, nullptr, nullptr, lora);
  // 2) Wt = (kron1 (x) kron2) @ QR, A generated on the fly
  hipLaunchKernelGGL((gemm_bf16x2<AMODE_KRON, BMODE_DIRECT>), g1, blk, 0, stream,
                     nullptr, QRbuf, Wtbuf, 4096, 4096, 4096, kron1, kron2, nullptr);
  // 3) out = x @ Wt   (SCALE == 1.0)
  hipLaunchKernelGGL((gemm_bf16x2<AMODE_DIRECT, BMODE_DIRECT>), g3, blk, 0, stream,
                     x, Wtbuf, out, 8192, 4096, 4096, nullptr, nullptr, nullptr);
}
